// ModelPointer_51049981280514
// MI455X (gfx1250) — compile-verified, matrix operand order checked by reading, not run on hardware
//
#include <hip/hip_runtime.h>
#include <math.h>

// ---------------------------------------------------------------------------
// Problem constants (match reference)
// ---------------------------------------------------------------------------
constexpr int BB  = 16;     // batch
constexpr int SS  = 512;    // sequence
constexpr int FD  = 512;    // model dim
constexpr int NH  = 8;      // heads
constexpr int HD  = 64;     // head dim
constexpr int FFD = 2048;   // ff dim
constexpr int NL  = 3;      // layers per phase
constexpr int NTOK = BB * SS;           // 8192 tokens
constexpr float LNEPS = 1e-5f;

typedef __bf16 bf16_t;
typedef __attribute__((ext_vector_type(16))) __bf16 v16bf;
typedef __attribute__((ext_vector_type(8)))  float  v8f;

// ---------------------------------------------------------------------------
// CDNA5 async global->LDS copy (GLOBAL_LOAD_ASYNC_TO_LDS_B128, ASYNCcnt).
// Builtin param 0 is (per hipcc diagnostic) a pointer to int32x4 in the
// global address space; param 1 is the LDS-side pointer. We materialize the
// address-space-qualified pointers via integer casts (always legal; low 32
// bits of a generic LDS address are the LDS byte offset per ISA 10.2).
// ---------------------------------------------------------------------------
#if defined(__HIP_DEVICE_COMPILE__) &&                                   \
    __has_builtin(__builtin_amdgcn_global_load_async_to_lds_b128) &&     \
    __has_builtin(__builtin_amdgcn_s_wait_asynccnt)
#define USE_ASYNC_LDS 1
#else
#define USE_ASYNC_LDS 0
#endif

#if USE_ASYNC_LDS
typedef int v4i __attribute__((vector_size(4 * sizeof(int))));
typedef __attribute__((address_space(1))) v4i v4i_as1;
typedef __attribute__((address_space(3))) v4i v4i_as3;

__device__ __forceinline__ void async_b128(const bf16_t* g, bf16_t* l) {
  v4i_as1* gp = (v4i_as1*)(unsigned long long)g;
  v4i_as3* lp = (v4i_as3*)(unsigned int)(unsigned long long)l;
  __builtin_amdgcn_global_load_async_to_lds_b128(gp, lp, 0, 0);
}

// copy 16 bf16 (32 bytes) per lane: two b128 async transfers
__device__ __forceinline__ void async_copy32B(const bf16_t* g, bf16_t* l) {
  async_b128(g, l);
  async_b128(g + 8, l + 8);
}
#endif

// ---------------------------------------------------------------------------
// WMMA helpers (wave32, v_wmma_f32_16x16x32_bf16)
// A-frag / B-frag (from an [rows][ld] bf16 matrix, row per lane):
//   lane l -> row (l & 15); halves: elems 0..7 at K = koff..koff+7,
//   elems 8..15 at K = 16+koff..16+koff+7, koff = (l>=16) ? 8 : 0.
// C/D frag: VGPR r, lane l -> (row = r + 8*(l>>4), col = l & 15).
// ---------------------------------------------------------------------------
__device__ __forceinline__ v16bf load_frag(const bf16_t* base, int ld) {
  const int lane = threadIdx.x & 31;
  const bf16_t* p = base + (size_t)(lane & 15) * ld + ((lane >> 4) << 3);
  v16bf f;
#pragma unroll
  for (int e = 0; e < 8; ++e) f[e] = p[e];
#pragma unroll
  for (int e = 0; e < 8; ++e) f[e + 8] = p[e + 16];
  return f;
}

__device__ __forceinline__ v8f wmma_bf16(v16bf a, v16bf b, v8f c) {
  return __builtin_amdgcn_wmma_f32_16x16x32_bf16(
      /*neg_a=*/false, a, /*neg_b=*/false, b,
      /*c_mod=*/(short)0, c, /*reuse_a=*/false, /*reuse_b=*/false);
}

__device__ __forceinline__ v8f zero8() {
  v8f z = {0.f, 0.f, 0.f, 0.f, 0.f, 0.f, 0.f, 0.f};
  return z;
}

__device__ __forceinline__ float gelu_exact(float x) {
  return 0.5f * x * (1.0f + erff(x * 0.70710678118654752f));
}

// ---------------------------------------------------------------------------
// Kernel 1: tiled transpose + f32->bf16 convert:  Wt[N][K] = (bf16)W[K][N]
// ---------------------------------------------------------------------------
__global__ __launch_bounds__(256) void transpose_convert_kernel(
    const float* __restrict__ W, bf16_t* __restrict__ Wt, int K, int N) {
  __shared__ float tile[32][33];
  const int bk = blockIdx.x * 32;
  const int bn = blockIdx.y * 32;
  const int tx = threadIdx.x & 31;
  const int ty = threadIdx.x >> 5;  // 0..7
#pragma unroll
  for (int i = 0; i < 32; i += 8)
    tile[ty + i][tx] = W[(size_t)(bk + ty + i) * N + bn + tx];
  __syncthreads();
#pragma unroll
  for (int i = 0; i < 32; i += 8)
    Wt[(size_t)(bn + ty + i) * K + bk + tx] = (bf16_t)tile[tx][ty + i];
}

// ---------------------------------------------------------------------------
// Kernel 2: f32 -> bf16 elementwise
// ---------------------------------------------------------------------------
__global__ __launch_bounds__(256) void f32_to_bf16_kernel(
    const float* __restrict__ in, bf16_t* __restrict__ out, size_t n) {
  size_t i = (size_t)blockIdx.x * 256 + threadIdx.x;
  if (i < n) out[i] = (bf16_t)in[i];
}

// ---------------------------------------------------------------------------
// Kernel 3: GEMM  C[M,N] = act(A[M,K] @ W[K,N] + bias[N])
//   A: bf16 [M][K] row-major.  Wt: bf16 [N][K] (pre-transposed weights).
//   mode: 0 = f32 out, 1 = bf16 out, 2 = bf16 out + exact GELU
//   Block: 256 thr = 8 waves; block tile 128x128; wave tile 64x32 (4x2 WMMA).
//   K-loop: double-buffered async global->LDS (ASYNCcnt) when available.
// ---------------------------------------------------------------------------
__global__ __launch_bounds__(256) void gemm_bf16_kernel(
    const bf16_t* __restrict__ A, const bf16_t* __restrict__ Wt,
    const float* __restrict__ bias, void* __restrict__ Cout,
    int M, int N, int Kdim, int mode) {
  constexpr int TK = 32;
  constexpr int LDT = TK + 8;  // LDS pad (8 bf16 = 16B)
#if USE_ASYNC_LDS
  __shared__ bf16_t As[2][128][LDT];
  __shared__ bf16_t Bs[2][128][LDT];
#else
  __shared__ bf16_t As[1][128][LDT];
  __shared__ bf16_t Bs[1][128][LDT];
#endif

  const int m0 = blockIdx.y * 128;
  const int n0 = blockIdx.x * 128;
  const int tid = threadIdx.x;
  const int wave = tid >> 5;
  const int wm = wave & 1;   // 2 waves along M (64 each)
  const int wn = wave >> 1;  // 4 waves along N (32 each)

  v8f acc[4][2];
#pragma unroll
  for (int i = 0; i < 4; ++i)
#pragma unroll
    for (int j = 0; j < 2; ++j) acc[i][j] = zero8();

  const int lr = tid >> 1;          // 0..127 : tile row handled by this thread
  const int lc = (tid & 1) * 16;    // 0 or 16 : 16-elem (32B) segment
  const int T = Kdim / TK;

#if USE_ASYNC_LDS
  auto issue_tile = [&](int t) {
    const int buf = t & 1;
    const bf16_t* ga = A  + (size_t)(m0 + lr) * Kdim + t * TK + lc;
    const bf16_t* gb = Wt + (size_t)(n0 + lr) * Kdim + t * TK + lc;
    async_copy32B(ga, &As[buf][lr][lc]);   // 2 async b128
    async_copy32B(gb, &Bs[buf][lr][lc]);   // 2 async b128
  };
  issue_tile(0);
#endif

  for (int t = 0; t < T; ++t) {
#if USE_ASYNC_LDS
    const int buf = t & 1;
    if (t + 1 < T) {
      issue_tile(t + 1);                       // tile t+1 in flight (4 ops)
      __builtin_amdgcn_s_wait_asynccnt(4);     // tile t fully landed in LDS
    } else {
      __builtin_amdgcn_s_wait_asynccnt(0);
    }
#else
    const int buf = 0;
    {
      const bf16_t* ap = A  + (size_t)(m0 + lr) * Kdim + t * TK + lc;
      const bf16_t* bp = Wt + (size_t)(n0 + lr) * Kdim + t * TK + lc;
#pragma unroll
      for (int e = 0; e < 16; ++e) As[0][lr][lc + e] = ap[e];
#pragma unroll
      for (int e = 0; e < 16; ++e) Bs[0][lr][lc + e] = bp[e];
      if (t + 1 < T) {
        __builtin_prefetch(ap + TK, 0, 3);
        __builtin_prefetch(bp + TK, 0, 3);
      }
    }
#endif
    __syncthreads();

    v16bf af[4], bfrag[2];
#pragma unroll
    for (int i = 0; i < 4; ++i)
      af[i] = load_frag(&As[buf][wm * 64 + i * 16][0], LDT);
#pragma unroll
    for (int j = 0; j < 2; ++j)
      bfrag[j] = load_frag(&Bs[buf][wn * 32 + j * 16][0], LDT);
#pragma unroll
    for (int i = 0; i < 4; ++i)
#pragma unroll
      for (int j = 0; j < 2; ++j)
        acc[i][j] = wmma_bf16(af[i], bfrag[j], acc[i][j]);

    __syncthreads();  // protect LDS buffer reuse
  }

  // Epilogue
  const int lane = tid & 31;
  const int colq = lane & 15;
  const int rowoff = (lane >> 4) * 8;
  float* Cf = (float*)Cout;
  bf16_t* Cb = (bf16_t*)Cout;
#pragma unroll
  for (int i = 0; i < 4; ++i)
#pragma unroll
    for (int j = 0; j < 2; ++j) {
      const int n = n0 + wn * 32 + j * 16 + colq;
      const float bv = bias[n];
#pragma unroll
      for (int r = 0; r < 8; ++r) {
        const size_t m = (size_t)m0 + wm * 64 + i * 16 + rowoff + r;
        float v = acc[i][j][r] + bv;
        if (mode == 2) v = gelu_exact(v);
        if (mode == 0) Cf[m * N + n] = v;
        else           Cb[m * N + n] = (bf16_t)v;
      }
    }
}

// ---------------------------------------------------------------------------
// Kernel 4: split heads  [B*S, F] -> Q/K [B,H,S,D] or V^T [B,H,D,S]
// ---------------------------------------------------------------------------
__global__ __launch_bounds__(256) void split_heads_kernel(
    const bf16_t* __restrict__ in, bf16_t* __restrict__ out, int vtrans) {
  const size_t idx = (size_t)blockIdx.x * 256 + threadIdx.x;  // B*S*F
  const int f = (int)(idx & (FD - 1));
  const size_t bs = idx >> 9;
  const int s = (int)(bs & (SS - 1));
  const int b = (int)(bs >> 9);
  const int h = f >> 6;
  const int d = f & (HD - 1);
  const bf16_t v = in[idx];
  if (vtrans)
    out[(((size_t)b * NH + h) * HD + d) * SS + s] = v;
  else
    out[(((size_t)b * NH + h) * SS + s) * HD + d] = v;
}

// ---------------------------------------------------------------------------
// Kernel 5: flash attention, one wave (32 thr) per (b, h, 16-query tile).
//   Q,K: bf16 [B,H,S,64]; Vt: bf16 [B,H,64,S]; out: bf16 [B,S,F].
//   mode: 0 = encoder (key < len), 1 = decoder (key < min(max(t,1), len)).
// ---------------------------------------------------------------------------
__global__ __launch_bounds__(32) void flash_attn_kernel(
    const bf16_t* __restrict__ Q, const bf16_t* __restrict__ K,
    const bf16_t* __restrict__ Vt, const int* __restrict__ lengths,
    bf16_t* __restrict__ out, int dec_mode) {
  __shared__ bf16_t plds[16][40];

  const int bid = blockIdx.x;
  const int qt = bid & (SS / 16 - 1);          // 0..31
  const int h  = (bid >> 5) & (NH - 1);
  const int b  = bid >> 8;
  const int lane = threadIdx.x & 31;
  const int col = lane & 15;
  const int rowoff = (lane >> 4) * 8;

  const bf16_t* Qb = Q  + (((size_t)b * NH + h) * SS + qt * 16) * HD;
  const bf16_t* Kb = K  + ((size_t)b * NH + h) * SS * HD;
  const bf16_t* Vb = Vt + ((size_t)b * NH + h) * HD * SS;

  const v16bf qf0 = load_frag(Qb, HD);        // dims 0..31
  const v16bf qf1 = load_frag(Qb + 32, HD);   // dims 32..63

  const int len = lengths[b];
  int limit[8];
#pragma unroll
  for (int r = 0; r < 8; ++r) {
    const int t = qt * 16 + rowoff + r;
    if (dec_mode) {
      int lim = (t < 1) ? 1 : t;
      limit[r] = (lim > len) ? len : lim;
    } else {
      limit[r] = len;
    }
  }

  v8f accv[4];
#pragma unroll
  for (int nt = 0; nt < 4; ++nt) accv[nt] = zero8();
  float m_i[8], l_i[8];
#pragma unroll
  for (int r = 0; r < 8; ++r) { m_i[r] = -1e30f; l_i[r] = 0.f; }

  for (int kb = 0; kb < SS; kb += 32) {
    // ---- scores: Q(16x64) . K^T(64x32) -> 2 col-tiles of 16 ----
    v8f sc[2];
#pragma unroll
    for (int ct = 0; ct < 2; ++ct) {
      const bf16_t* kp = Kb + (size_t)(kb + ct * 16) * HD;
      v16bf k0 = load_frag(kp, HD);
      v16bf k1 = load_frag(kp + 32, HD);
      v8f c = zero8();
      c = wmma_bf16(qf0, k0, c);
      c = wmma_bf16(qf1, k1, c);
      sc[ct] = c;
    }
    // ---- scale + mask ----
#pragma unroll
    for (int ct = 0; ct < 2; ++ct)
#pragma unroll
      for (int r = 0; r < 8; ++r) {
        float v = sc[ct][r] * 0.125f;  // 1/sqrt(64)
        const int key = kb + ct * 16 + col;
        if (key >= limit[r]) v = -1e30f;
        sc[ct][r] = v;
      }
    // ---- online softmax over this 32-key chunk ----
#pragma unroll
    for (int r = 0; r < 8; ++r) {
      float cm = fmaxf(sc[0][r], sc[1][r]);
      cm = fmaxf(cm, __shfl_xor(cm, 1));
      cm = fmaxf(cm, __shfl_xor(cm, 2));
      cm = fmaxf(cm, __shfl_xor(cm, 4));
      cm = fmaxf(cm, __shfl_xor(cm, 8));
      const float mn = fmaxf(m_i[r], cm);
      const float corr = __expf(m_i[r] - mn);
      const float p0 = __expf(sc[0][r] - mn);
      const float p1 = __expf(sc[1][r] - mn);
      sc[0][r] = p0;
      sc[1][r] = p1;
      float rs = p0 + p1;
      rs += __shfl_xor(rs, 1);
      rs += __shfl_xor(rs, 2);
      rs += __shfl_xor(rs, 4);
      rs += __shfl_xor(rs, 8);
      l_i[r] = l_i[r] * corr + rs;
      m_i[r] = mn;
#pragma unroll
      for (int nt = 0; nt < 4; ++nt) accv[nt][r] *= corr;
    }
    // ---- P (C-frag layout) -> LDS -> A-frag layout ----
#pragma unroll
    for (int ct = 0; ct < 2; ++ct)
#pragma unroll
      for (int r = 0; r < 8; ++r)
        plds[rowoff + r][ct * 16 + col] = (bf16_t)sc[ct][r];
    __syncthreads();
    const v16bf pa = load_frag(&plds[0][0], 40);
    // ---- O += P(16x32) . V(32x64) ----
#pragma unroll
    for (int nt = 0; nt < 4; ++nt) {
      v16bf vf = load_frag(Vb + (size_t)(nt * 16) * SS + kb, SS);
      accv[nt] = wmma_bf16(pa, vf, accv[nt]);
    }
    __syncthreads();
  }

  // ---- normalize and merge heads into [B,S,F] bf16 ----
  bf16_t* ob = out + ((size_t)b * SS + qt * 16) * FD + h * HD;
#pragma unroll
  for (int nt = 0; nt < 4; ++nt)
#pragma unroll
    for (int r = 0; r < 8; ++r)
      ob[(size_t)(rowoff + r) * FD + nt * 16 + col] =
          (bf16_t)(accv[nt][r] / l_i[r]);
}

// ---------------------------------------------------------------------------
// Kernel 6: residual add + LayerNorm; writes f32 master (in place) + bf16 copy
// ---------------------------------------------------------------------------
__global__ __launch_bounds__(256) void add_ln_kernel(
    float* __restrict__ x, const float* __restrict__ delta,
    const float* __restrict__ g, const float* __restrict__ beta,
    bf16_t* __restrict__ xbf) {
  const size_t row = blockIdx.x;
  const int tid = threadIdx.x;
  const size_t base = row * FD;
  const float v0 = x[base + tid]       + delta[base + tid];
  const float v1 = x[base + 256 + tid] + delta[base + 256 + tid];

  float s = v0 + v1;
  float s2 = v0 * v0 + v1 * v1;
#pragma unroll
  for (int o = 16; o > 0; o >>= 1) {
    s  += __shfl_xor(s,  o);
    s2 += __shfl_xor(s2, o);
  }
  __shared__ float ps[8], ps2[8];
  const int w = tid >> 5;
  if ((tid & 31) == 0) { ps[w] = s; ps2[w] = s2; }
  __syncthreads();
  if (tid == 0) {
    float a = 0.f, b2 = 0.f;
#pragma unroll
    for (int i = 0; i < 8; ++i) { a += ps[i]; b2 += ps2[i]; }
    ps[0] = a; ps2[0] = b2;
  }
  __syncthreads();
  const float mean = ps[0] * (1.0f / FD);
  const float var  = ps2[0] * (1.0f / FD) - mean * mean;
  const float inv  = rsqrtf(var + LNEPS);

  const float o0 = (v0 - mean) * inv * g[tid] + beta[tid];
  const float o1 = (v1 - mean) * inv * g[256 + tid] + beta[256 + tid];
  x[base + tid] = o0;
  x[base + 256 + tid] = o1;
  xbf[base + tid] = (bf16_t)o0;
  xbf[base + 256 + tid] = (bf16_t)o1;
}

// ---------------------------------------------------------------------------
// Kernel 7: row dot:  out[row] = X[row,:] . w   (512 elems, one wave per row)
// ---------------------------------------------------------------------------
__global__ __launch_bounds__(256) void rowdot_kernel(
    const float* __restrict__ X, const float* __restrict__ w,
    float* __restrict__ out) {
  const size_t row = (size_t)blockIdx.x * 8 + (threadIdx.x >> 5);
  const int lane = threadIdx.x & 31;
  float s = 0.f;
#pragma unroll
  for (int f = 0; f < FD / 32; ++f)
    s += X[row * FD + f * 32 + lane] * w[f * 32 + lane];
#pragma unroll
  for (int o = 16; o > 0; o >>= 1) s += __shfl_xor(s, o);
  if (lane == 0) out[row] = s;
}

// ---------------------------------------------------------------------------
// Kernel 8: pointer output
//   out[b,i,j] = (j==0) ? 0 : sigmoid((wq[b,j] + wk[b,i]) / sqrt(F))
// ---------------------------------------------------------------------------
__global__ __launch_bounds__(256) void pointer_out_kernel(
    const float* __restrict__ wq, const float* __restrict__ wk,
    float* __restrict__ out) {
  const size_t idx = (size_t)blockIdx.x * 256 + threadIdx.x;  // B*S*S
  if (idx >= (size_t)BB * SS * SS) return;
  const int j = (int)(idx & (SS - 1));
  const size_t bi = idx >> 9;
  const int i = (int)(bi & (SS - 1));
  const int b = (int)(bi >> 9);
  float v = 0.f;
  if (j != 0) {
    const float z = (wq[(size_t)b * SS + j] + wk[(size_t)b * SS + i]) *
                    0.04419417382415922f;  // 1/sqrt(512)
    v = 1.f / (1.f + __expf(-z));
  }
  out[idx] = v;
}

// ---------------------------------------------------------------------------
// Host orchestration
// ---------------------------------------------------------------------------
extern "C" void kernel_launch(void* const* d_in, const int* in_sizes, int n_in,
                              void* d_out, int out_size, void* d_ws, size_t ws_size,
                              hipStream_t stream) {
  (void)in_sizes; (void)n_in; (void)out_size; (void)ws_size;

  const float* emb     = (const float*)d_in[0];
  const int*   lengths = (const int*)d_in[1];
  const float* ptr_w_q = (const float*)d_in[26];
  const float* ptr_w_k = (const float*)d_in[27];

  const size_t F2 = (size_t)FD * FD;                       // 262144
  const size_t per_layer = 4 * F2 + 2 * (size_t)FD * FFD;  // 3,145,728 elems

  // ---- carve workspace ----
  char* wsp = (char*)d_ws;
  auto carve = [&](size_t bytes) -> void* {
    void* r = (void*)wsp;
    wsp += (bytes + 255) & ~(size_t)255;
    return r;
  };
  bf16_t* WT     = (bf16_t*)carve(6 * per_layer * sizeof(bf16_t));
  float*  xf     = (float*) carve((size_t)NTOK * FD * 4);
  bf16_t* xbf    = (bf16_t*)carve((size_t)NTOK * FD * 2);
  float*  memf   = (float*) carve((size_t)NTOK * FD * 4);
  bf16_t* membf  = (bf16_t*)carve((size_t)NTOK * FD * 2);
  float*  tmpf   = (float*) carve((size_t)NTOK * FD * 4);
  bf16_t* qkvtmp = (bf16_t*)carve((size_t)NTOK * FD * 2);
  bf16_t* Qb     = (bf16_t*)carve((size_t)NTOK * FD * 2);
  bf16_t* Kb     = (bf16_t*)carve((size_t)NTOK * FD * 2);
  bf16_t* Vt     = (bf16_t*)carve((size_t)NTOK * FD * 2);
  bf16_t* attnb  = (bf16_t*)carve((size_t)NTOK * FD * 2);
  bf16_t* h1     = (bf16_t*)carve((size_t)NTOK * FFD * 2);
  float*  wqv    = (float*) carve((size_t)NTOK * 4);
  float*  wkv    = (float*) carve((size_t)NTOK * 4);

  // ---- weight transpose + bf16 convert (per call; deterministic) ----
  for (int p = 0; p < 2; ++p) {
    const int bi = 2 + 12 * p;
    const float* wqkv = (const float*)d_in[bi + 0];
    const float* wo   = (const float*)d_in[bi + 2];
    const float* w1   = (const float*)d_in[bi + 6];
    const float* w2   = (const float*)d_in[bi + 8];
    for (int l = 0; l < NL; ++l) {
      bf16_t* base = WT + ((size_t)p * NL + l) * per_layer;
      for (int j = 0; j < 3; ++j)
        transpose_convert_kernel<<<dim3(FD / 32, FD / 32), 256, 0, stream>>>(
            wqkv + ((size_t)l * 3 + j) * F2, base + j * F2, FD, FD);
      transpose_convert_kernel<<<dim3(FD / 32, FD / 32), 256, 0, stream>>>(
          wo + (size_t)l * F2, base + 3 * F2, FD, FD);
      transpose_convert_kernel<<<dim3(FD / 32, FFD / 32), 256, 0, stream>>>(
          w1 + (size_t)l * FD * FFD, base + 4 * F2, FD, FFD);
      transpose_convert_kernel<<<dim3(FFD / 32, FD / 32), 256, 0, stream>>>(
          w2 + (size_t)l * FFD * FD, base + 4 * F2 + (size_t)FD * FFD, FFD, FD);
    }
  }

  const size_t NTOT = (size_t)NTOK * FD;           // 4,194,304
  const int    CVB  = (int)(NTOT / 256);           // 16384
  const dim3   G512(FD / 128, NTOK / 128);         // (4, 64)
  const dim3   GFF1(FFD / 128, NTOK / 128);        // (16, 64)

  auto run_layer = [&](int p, int l, const bf16_t* q_src, const bf16_t* kv_src) {
    const int bi = 2 + 12 * p;
    const float* bqkv = (const float*)d_in[bi + 1];
    const float* bo   = (const float*)d_in[bi + 3];
    const float* ln1g = (const float*)d_in[bi + 4];
    const float* ln1b = (const float*)d_in[bi + 5];
    const float* b1   = (const float*)d_in[bi + 7];
    const float* b2   = (const float*)d_in[bi + 9];
    const float* ln2g = (const float*)d_in[bi + 10];
    const float* ln2b = (const float*)d_in[bi + 11];
    bf16_t* base = WT + ((size_t)p * NL + l) * per_layer;

    // Q / K / V projections (bf16 out), head split (V transposed)
    gemm_bf16_kernel<<<G512, 256, 0, stream>>>(
        q_src, base + 0 * F2, bqkv + ((size_t)l * 3 + 0) * FD, qkvtmp,
        NTOK, FD, FD, 1);
    split_heads_kernel<<<CVB, 256, 0, stream>>>(qkvtmp, Qb, 0);
    gemm_bf16_kernel<<<G512, 256, 0, stream>>>(
        kv_src, base + 1 * F2, bqkv + ((size_t)l * 3 + 1) * FD, qkvtmp,
        NTOK, FD, FD, 1);
    split_heads_kernel<<<CVB, 256, 0, stream>>>(qkvtmp, Kb, 0);
    gemm_bf16_kernel<<<G512, 256, 0, stream>>>(
        kv_src, base + 2 * F2, bqkv + ((size_t)l * 3 + 2) * FD, qkvtmp,
        NTOK, FD, FD, 1);
    split_heads_kernel<<<CVB, 256, 0, stream>>>(qkvtmp, Vt, 1);

    flash_attn_kernel<<<BB * NH * (SS / 16), 32, 0, stream>>>(
        Qb, Kb, Vt, lengths, attnb, p /* dec mode when p==1 */);

    // output projection (f32), residual + LN1
    gemm_bf16_kernel<<<G512, 256, 0, stream>>>(
        attnb, base + 3 * F2, bo + (size_t)l * FD, tmpf, NTOK, FD, FD, 0);
    add_ln_kernel<<<NTOK, 256, 0, stream>>>(
        xf, tmpf, ln1g + (size_t)l * FD, ln1b + (size_t)l * FD, xbf);

    // FFN: GELU(x W1 + b1) W2 + b2, residual + LN2
    gemm_bf16_kernel<<<GFF1, 256, 0, stream>>>(
        xbf, base + 4 * F2, b1 + (size_t)l * FFD, h1, NTOK, FFD, FD, 2);
    gemm_bf16_kernel<<<G512, 256, 0, stream>>>(
        h1, base + 4 * F2 + (size_t)FD * FFD, b2 + (size_t)l * FD, tmpf,
        NTOK, FD, FFD, 0);
    add_ln_kernel<<<NTOK, 256, 0, stream>>>(
        xf, tmpf, ln2g + (size_t)l * FD, ln2b + (size_t)l * FD, xbf);
  };

  // ---- encoder ----
  (void)hipMemcpyAsync(xf, emb, NTOT * 4, hipMemcpyDeviceToDevice, stream);
  f32_to_bf16_kernel<<<CVB, 256, 0, stream>>>(xf, xbf, NTOT);
  for (int l = 0; l < NL; ++l) run_layer(0, l, xbf, xbf);

  // snapshot encoder memory
  (void)hipMemcpyAsync(memf,  xf,  NTOT * 4, hipMemcpyDeviceToDevice, stream);
  (void)hipMemcpyAsync(membf, xbf, NTOT * 2, hipMemcpyDeviceToDevice, stream);

  // ---- decoder (cross-attends to memory) ----
  (void)hipMemcpyAsync(xf, emb, NTOT * 4, hipMemcpyDeviceToDevice, stream);
  f32_to_bf16_kernel<<<CVB, 256, 0, stream>>>(xf, xbf, NTOT);
  for (int l = 0; l < NL; ++l) run_layer(1, l, xbf, membf);

  // ---- pointer head ----
  rowdot_kernel<<<NTOK / 8, 256, 0, stream>>>(xf,   ptr_w_q, wqv);
  rowdot_kernel<<<NTOK / 8, 256, 0, stream>>>(memf, ptr_w_k, wkv);
  pointer_out_kernel<<<(int)(((size_t)BB * SS * SS) / 256), 256, 0, stream>>>(
      wqv, wkv, (float*)d_out);
}
